// SplitAttention_11098195493013
// MI455X (gfx1250) — matrix operand order repaired and checked by reading, not compiled, not run on hardware
//
#include <hip/hip_runtime.h>
#include <hip/hip_bf16.h>

// ---------------------------------------------------------------------------
// SplitAttention for MI455X (gfx1250): everything through v_wmma_f32_16x16x32_f16.
// B=8, N=1024, C=1024, H=16, HD=64.
// ---------------------------------------------------------------------------

typedef _Float16 half_t;
typedef __attribute__((ext_vector_type(16))) _Float16 v16h;
typedef __attribute__((ext_vector_type(4)))  _Float16 v4h;
typedef __attribute__((ext_vector_type(8)))  float    v8f;

#define BB 8
#define NN 1024
#define CC 1024
#define HH 16
#define HD 64
#define MTOT (BB * NN)       // 8192 rows
#define ATT_SCALE 0.125f     // 1/sqrt(64)

union FragH {
    v16h h;
    uint4 q[2];
};

// ---------------------------------------------------------------------------
// fp32 -> f16 conversion (vectorized by 4)
// ---------------------------------------------------------------------------
__global__ void cvt_f32_to_f16(const float* __restrict__ in,
                               half_t* __restrict__ out, int n4) {
    int i = blockIdx.x * blockDim.x + threadIdx.x;
    if (i < n4) {
        float4 v = reinterpret_cast<const float4*>(in)[i];
        v4h o = { (half_t)v.x, (half_t)v.y, (half_t)v.z, (half_t)v.w };
        reinterpret_cast<v4h*>(out)[i] = o;
    }
}

// ---------------------------------------------------------------------------
// WMMA GEMM: D[m,n] = sum_k A[m,k] * Wt[n,k]      (torch Linear: x @ W^T)
// A: [8192 x 1024] f16 row-major, Wt: [1024 x 1024] f16 row-major.
// Block = 256 threads (8 waves). Block tile 256x128, wave tile 64x64:
// per K-step of 32: 4 A-frags + 4 B-frags (16 x B128) feed 16 WMMAs
// (1.0 vmem/wmma). A + Wt are L2-resident (192 MB), so no LDS staging.
// MODE 0: f16 row-major out.  MODE 1: f16 scattered into Vt[b,h,d,tok].
// MODE 2: f32 out + bias (final projection).
// ---------------------------------------------------------------------------
template <int MODE>
__global__ void __launch_bounds__(256)
gemm_wmma(const half_t* __restrict__ A, const half_t* __restrict__ Wt,
          void* __restrict__ out, const float* __restrict__ bias) {
    const int K = CC;
    const int lane = threadIdx.x & 31;
    const int wave = threadIdx.x >> 5;
    const int l15  = lane & 15;
    const int hi   = (lane >> 4) & 1;       // 0: lanes 0-15, 1: lanes 16-31
    const int aK0  = hi * 8;                // A-frag K base within lane
    const int bK0  = hi * 8;                // B-frag K base within lane
    const int hi8  = hi * 8;                // D-layout row offset

    const int wm = wave & 3;                // 4 waves along M
    const int wn = wave >> 2;               // 2 waves along N
    const int mBase = blockIdx.y * 256 + wm * 64;
    const int nBase = blockIdx.x * 128 + wn * 64;

    v8f acc[4][4];
#pragma unroll
    for (int mi = 0; mi < 4; ++mi)
#pragma unroll
        for (int ni = 0; ni < 4; ++ni)
            acc[mi][ni] = (v8f){};

    for (int k = 0; k < K; k += 32) {
        FragH a[4];
#pragma unroll
        for (int mi = 0; mi < 4; ++mi) {
            const half_t* p = A + (size_t)(mBase + mi * 16 + l15) * K + k + aK0;
            a[mi].q[0] = *reinterpret_cast<const uint4*>(p);
            a[mi].q[1] = *reinterpret_cast<const uint4*>(p + 16);
        }
        FragH bfr[4];
#pragma unroll
        for (int ni = 0; ni < 4; ++ni) {
            const half_t* p = Wt + (size_t)(nBase + ni * 16 + l15) * K + k + bK0;
            bfr[ni].q[0] = *reinterpret_cast<const uint4*>(p);
            bfr[ni].q[1] = *reinterpret_cast<const uint4*>(p + 16);
        }
#pragma unroll
        for (int mi = 0; mi < 4; ++mi)
#pragma unroll
            for (int ni = 0; ni < 4; ++ni)
                acc[mi][ni] = __builtin_amdgcn_wmma_f32_16x16x32_f16(
                    false, a[mi].h, false, bfr[ni].h,
                    (short)0, acc[mi][ni], false, false);
    }

    // D layout: VGPR r holds row M = r + hi*8, column N = lane&15.
#pragma unroll
    for (int mi = 0; mi < 4; ++mi) {
#pragma unroll
        for (int ni = 0; ni < 4; ++ni) {
#pragma unroll
            for (int r = 0; r < 8; ++r) {
                const int row = mBase + mi * 16 + r + hi8;
                const int col = nBase + ni * 16 + l15;
                const float v = acc[mi][ni][r];
                if (MODE == 0) {
                    reinterpret_cast<half_t*>(out)[(size_t)row * CC + col] = (half_t)v;
                } else if (MODE == 1) {
                    // Vt[b][h][d][tok]
                    const int b = row >> 10, tok = row & 1023;
                    const int h = col >> 6, d = col & 63;
                    reinterpret_cast<half_t*>(out)
                        [(((size_t)(b * HH + h)) * HD + d) * NN + tok] = (half_t)v;
                } else {
                    reinterpret_cast<float*>(out)[(size_t)row * CC + col] =
                        v + bias[col];
                }
            }
        }
    }
}

// ---------------------------------------------------------------------------
// Flash attention: one wave per (b, h, 16-row q tile), streaming 64 keys/step
// (4 S-tiles, 16 WMMAs per step) so the online-softmax shuffle/exp overhead
// is amortized over twice the keys.
// Q,K: f16 [tok][c] row-major; Vt: f16 [b][h][d][tok]; mask: [b][tok] u8.
// Output Ah: f16 [tok][c].
// ---------------------------------------------------------------------------
__global__ void __launch_bounds__(128)
flash_attn(const half_t* __restrict__ Qh, const half_t* __restrict__ Kh,
           const half_t* __restrict__ Vt, const unsigned char* __restrict__ mask,
           half_t* __restrict__ Ah) {
    __shared__ half_t ldsP[4][16][64];      // per-wave P tile (D-layout -> A-layout)

    const int lane = threadIdx.x & 31;
    const int wave = threadIdx.x >> 5;
    const int l15  = lane & 15;
    const int hi   = (lane >> 4) & 1;
    const int aK0  = hi * 8;
    const int bK0  = hi * 8;
    const int hi8  = hi * 8;

    const int bh = blockIdx.x;              // 0..127
    const int b  = bh >> 4;
    const int h  = bh & 15;
    const int qt = blockIdx.y * 4 + wave;   // 0..63
    const int qRowBase = qt * 16;

    // Q A-fragments, d = [0,32) and [32,64); loaded once, L2-resident.
    FragH qa[2];
#pragma unroll
    for (int ds = 0; ds < 2; ++ds) {
        const half_t* p = Qh + (size_t)(b * NN + qRowBase + l15) * CC
                             + h * HD + ds * 32 + aK0;
        qa[ds].q[0] = *reinterpret_cast<const uint4*>(p);
        qa[ds].q[1] = *reinterpret_cast<const uint4*>(p + 16);
    }

    const half_t* Vt_bh = Vt + (size_t)bh * HD * NN;  // [d][tok]
    const unsigned char* mk = mask + b * NN;

    float mstat[8], lstat[8];
#pragma unroll
    for (int r = 0; r < 8; ++r) { mstat[r] = -3.0e38f; lstat[r] = 0.0f; }
    v8f o[4];
#pragma unroll
    for (int dt = 0; dt < 4; ++dt) o[dt] = (v8f){};

    for (int kt = 0; kt < NN; kt += 64) {
        // ---- S = Q K^T for 16 rows x 64 keys (four 16x16 tiles) ----
        v8f s[4];
#pragma unroll
        for (int t = 0; t < 4; ++t) s[t] = (v8f){};
#pragma unroll
        for (int t = 0; t < 4; ++t) {
#pragma unroll
            for (int ds = 0; ds < 2; ++ds) {
                FragH bk;
                const half_t* p = Kh + (size_t)(b * NN + kt + t * 16 + l15) * CC
                                     + h * HD + ds * 32 + bK0;
                bk.q[0] = *reinterpret_cast<const uint4*>(p);
                bk.q[1] = *reinterpret_cast<const uint4*>(p + 16);
                s[t] = __builtin_amdgcn_wmma_f32_16x16x32_f16(
                    false, qa[ds].h, false, bk.h, (short)0, s[t], false, false);
            }
        }

        bool msk[4];
#pragma unroll
        for (int t = 0; t < 4; ++t) msk[t] = mk[kt + t * 16 + l15] != 0;

        // ---- online softmax over this 64-key slab ----
        float corr[8];
#pragma unroll
        for (int r = 0; r < 8; ++r) {
            float v[4];
#pragma unroll
            for (int t = 0; t < 4; ++t)
                v[t] = msk[t] ? -1.0e30f : s[t][r] * ATT_SCALE;
            float tmax = fmaxf(fmaxf(v[0], v[1]), fmaxf(v[2], v[3]));
            tmax = fmaxf(tmax, __shfl_xor(tmax, 1, 32));
            tmax = fmaxf(tmax, __shfl_xor(tmax, 2, 32));
            tmax = fmaxf(tmax, __shfl_xor(tmax, 4, 32));
            tmax = fmaxf(tmax, __shfl_xor(tmax, 8, 32)); // row max per 16-lane half
            const float mnew = fmaxf(mstat[r], tmax);
            float p[4];
#pragma unroll
            for (int t = 0; t < 4; ++t) p[t] = __expf(v[t] - mnew);
            float ps = (p[0] + p[1]) + (p[2] + p[3]);
            ps += __shfl_xor(ps, 1, 32);
            ps += __shfl_xor(ps, 2, 32);
            ps += __shfl_xor(ps, 4, 32);
            ps += __shfl_xor(ps, 8, 32);
            corr[r]  = __expf(mstat[r] - mnew);
            lstat[r] = lstat[r] * corr[r] + ps;
            mstat[r] = mnew;
            // store P row-major [row][key_local] for the A-layout reload
#pragma unroll
            for (int t = 0; t < 4; ++t)
                ldsP[wave][r + hi8][t * 16 + l15] = (half_t)p[t];
        }

        // rescale running output
#pragma unroll
        for (int dt = 0; dt < 4; ++dt)
#pragma unroll
            for (int r = 0; r < 8; ++r)
                o[dt][r] *= corr[r];

        // per-wave LDS region: just drain DS ops, no block barrier needed
        asm volatile("s_wait_dscnt 0" ::: "memory");

        // ---- O += P * V over the two 32-key groups ----
#pragma unroll
        for (int g = 0; g < 2; ++g) {
            FragH pf;   // P as A-fragment (16x32, K = key_local in this group)
            {
                const half_t* p = &ldsP[wave][l15][g * 32 + aK0];
                pf.q[0] = *reinterpret_cast<const uint4*>(p);
                pf.q[1] = *reinterpret_cast<const uint4*>(p + 16);
            }
#pragma unroll
            for (int dt = 0; dt < 4; ++dt) {
                FragH bv;   // B[k=key][n=d] = Vt[d][key], contiguous over key
                const half_t* p = Vt_bh + (size_t)(dt * 16 + l15) * NN
                                        + kt + g * 32 + bK0;
                bv.q[0] = *reinterpret_cast<const uint4*>(p);
                bv.q[1] = *reinterpret_cast<const uint4*>(p + 16);
                o[dt] = __builtin_amdgcn_wmma_f32_16x16x32_f16(
                    false, pf.h, false, bv.h, (short)0, o[dt], false, false);
            }
        }
    }

    // epilogue: normalize and emit f16 [tok][c]
#pragma unroll
    for (int r = 0; r < 8; ++r) {
        const float inv = lstat[r] > 0.0f ? 1.0f / lstat[r] : 0.0f;
        const int tok = qRowBase + r + hi8;
#pragma unroll
        for (int dt = 0; dt < 4; ++dt) {
            Ah[(size_t)(b * NN + tok) * CC + h * HD + dt * 16 + l15] =
                (half_t)(o[dt][r] * inv);
        }
    }
}

// ---------------------------------------------------------------------------
// Host-side launcher
// ---------------------------------------------------------------------------
extern "C" void kernel_launch(void* const* d_in, const int* in_sizes, int n_in,
                              void* d_out, int out_size, void* d_ws, size_t ws_size,
                              hipStream_t stream) {
    const float* x  = (const float*)d_in[0];
    const unsigned char* mask = (const unsigned char*)d_in[1]; // jnp bool: 1 byte
    const float* Wq = (const float*)d_in[2];
    const float* Wk = (const float*)d_in[3];
    const float* Wv = (const float*)d_in[4];
    const float* Wp = (const float*)d_in[5];
    const float* bp = (const float*)d_in[6];
    float* out = (float*)d_out;

    const size_t MB = 1u << 20;
    char* ws = (char*)d_ws;
    half_t* Wq_h = (half_t*)(ws + 0 * MB);   //  2 MB each
    half_t* Wk_h = (half_t*)(ws + 2 * MB);
    half_t* Wv_h = (half_t*)(ws + 4 * MB);
    half_t* Wp_h = (half_t*)(ws + 6 * MB);
    half_t* Xh   = (half_t*)(ws + 8 * MB);   // 16 MB
    half_t* Qh   = (half_t*)(ws + 24 * MB);  // 16 MB
    half_t* Kh   = (half_t*)(ws + 40 * MB);  // 16 MB
    half_t* Vt   = (half_t*)(ws + 56 * MB);  // 16 MB  [b][h][d][tok]
    half_t* Ah   = (half_t*)(ws + 72 * MB);  // 16 MB  (needs ws_size >= 88 MB)

    // 1) fp32 -> f16 conversions
    {
        const int nx4 = MTOT * CC / 4;
        cvt_f32_to_f16<<<(nx4 + 255) / 256, 256, 0, stream>>>(x, Xh, nx4);
        const int nw4 = CC * CC / 4;
        cvt_f32_to_f16<<<(nw4 + 255) / 256, 256, 0, stream>>>(Wq, Wq_h, nw4);
        cvt_f32_to_f16<<<(nw4 + 255) / 256, 256, 0, stream>>>(Wk, Wk_h, nw4);
        cvt_f32_to_f16<<<(nw4 + 255) / 256, 256, 0, stream>>>(Wv, Wv_h, nw4);
        cvt_f32_to_f16<<<(nw4 + 255) / 256, 256, 0, stream>>>(Wp, Wp_h, nw4);
    }

    // 2) Q/K/V projections (WMMA GEMM, x @ W^T)
    dim3 ggrid(CC / 128, MTOT / 256);        // (8, 32)
    gemm_wmma<0><<<ggrid, 256, 0, stream>>>(Xh, Wq_h, (void*)Qh, nullptr);
    gemm_wmma<0><<<ggrid, 256, 0, stream>>>(Xh, Wk_h, (void*)Kh, nullptr);
    gemm_wmma<1><<<ggrid, 256, 0, stream>>>(Xh, Wv_h, (void*)Vt, nullptr);

    // 3) masked flash attention
    dim3 agrid(BB * HH, NN / 16 / 4);        // (128, 16), 4 waves/block
    flash_attn<<<agrid, 128, 0, stream>>>(Qh, Kh, Vt, mask, Ah);

    // 4) output projection + bias (f32 out)
    gemm_wmma<2><<<ggrid, 256, 0, stream>>>(Ah, Wp_h, (void*)out, bp);
}
